// SSMKernel_66907000537454
// MI455X (gfx1250) — compile-verified, matrix operand order checked by reading, not run on hardware
//
#include <hip/hip_runtime.h>
#include <math.h>

// ---------------------------------------------------------------------------
// Types
// ---------------------------------------------------------------------------
typedef __attribute__((ext_vector_type(16))) __bf16        v16bf;
typedef __attribute__((ext_vector_type(8)))  float         v8f;
typedef __attribute__((ext_vector_type(4)))  unsigned int  vu4;
typedef __attribute__((ext_vector_type(4)))  float         vf4;
typedef unsigned short u16t;

#define D_MODEL 2048
#define DT_RANK 64
#define M_ROWS  8192   // 4 * 2048

// ---------------------------------------------------------------------------
// bf16 helpers
// ---------------------------------------------------------------------------
// Full RNE (used in cheap, memory-bound prep kernels)
__device__ __forceinline__ unsigned short f2bf(float f) {
  unsigned int u = __float_as_uint(f);
  u += 0x7FFFu + ((u >> 16) & 1u);
  return (unsigned short)(u >> 16);
}
// Fast pack: round-to-nearest (ties up) + v_perm_b32 byte select: 3 VALU/pair
__device__ __forceinline__ unsigned int packbf2(float lo, float hi) {
  unsigned int a = __float_as_uint(lo) + 0x8000u;
  unsigned int b = __float_as_uint(hi) + 0x8000u;
  // result bytes: {b[3],b[2],a[3],a[2]}  ({src0,src1} = bytes 7..4, 3..0)
  return __builtin_amdgcn_perm(b, a, 0x07060302u);
}

union FragCast { vu4 u[2]; v16bf v; };

// Load one 16x32 bf16 WMMA A/B fragment from a row-major LDS tile.
// ISA 16-bit A layout: lanes 0-15 take K {0..7,16..23}, lanes 16-31 take
// K {8..15,24..31} of row (lane&15)+subtile_row. Two ds_load_b128 per frag.
__device__ __forceinline__ v16bf load_frag(const u16t* base, int row, int stride,
                                           int lane, int kofs) {
  const u16t* p = base + row * stride + kofs + (((lane >> 4) & 1) << 3);
  FragCast t;
  t.u[0] = *(const vu4*)(p);
  t.u[1] = *(const vu4*)(p + 16);
  return t.v;
}

__device__ __forceinline__ v8f wmma_bf16(v16bf a, v16bf b, v8f c) {
  return __builtin_amdgcn_wmma_f32_16x16x32_bf16(false, a, false, b,
                                                 (short)0, c, false, false);
}
__device__ __forceinline__ v8f vzero8() {
  v8f z;
#pragma unroll
  for (int i = 0; i < 8; ++i) z[i] = 0.0f;
  return z;
}

// ---------------------------------------------------------------------------
// Kernel 1: elementwise fp32 -> bf16 conversion (W1, W2)
// ---------------------------------------------------------------------------
__global__ void SSM_conv_bf16(const float* __restrict__ src,
                              u16t* __restrict__ dst, int n) {
  int i = blockIdx.x * 256 + threadIdx.x;
  if (i < n) dst[i] = f2bf(src[i]);
}

// ---------------------------------------------------------------------------
// Kernel 2: St[s][d] = bf16(A[d][s] + B[d][s] + C[d][s])  (LDS transpose)
// grid (64,64), block (32,8)
// ---------------------------------------------------------------------------
__global__ void SSM_sum_transpose(const float* __restrict__ A,
                                  const float* __restrict__ B,
                                  const float* __restrict__ C,
                                  u16t* __restrict__ St) {
  __shared__ float tile[32][33];
  const int bx = blockIdx.x * 32;  // s tile
  const int by = blockIdx.y * 32;  // d tile
  const int tx = threadIdx.x, ty = threadIdx.y;
#pragma unroll
  for (int j = 0; j < 32; j += 8) {
    size_t idx = (size_t)(by + ty + j) * D_MODEL + (bx + tx);
    tile[ty + j][tx] = A[idx] + B[idx] + C[idx];
  }
  __syncthreads();
#pragma unroll
  for (int j = 0; j < 32; j += 8) {
    St[(size_t)(bx + ty + j) * D_MODEL + (by + tx)] = f2bf(tile[tx][ty + j]);
  }
}

// ---------------------------------------------------------------------------
// Kernel 3: h = bf16(gelu(x @ W1^T + b1))   M=8192, N=64, K=2048
// tile 128x64, 256 threads = 8 waves (4x2), wave = 32x32 (2x2 WMMA subtiles)
// ping-pong LDS, one barrier per k-step
// ---------------------------------------------------------------------------
__global__ __launch_bounds__(256) void SSM_h_gemm(
    const float* __restrict__ x,     // [8192][2048] fp32
    const u16t*  __restrict__ W1b,   // [64][2048]   bf16 (N-major, K contig)
    const float* __restrict__ b1,    // [64]
    u16t*        __restrict__ hout)  // [8192][64]   bf16
{
  __shared__ __align__(16) u16t smem[2 * 6144];  // per buf: As 128x32, Ws 64x32

  const int tid  = threadIdx.x;
  const int lane = tid & 31;
  const int wm   = (tid >> 5) >> 1;  // 0..3
  const int wn   = (tid >> 5) & 1;   // 0..1
  const int m0   = blockIdx.x * 128;

  const int row2 = tid >> 1;          // 0..127 (x loader)
  const int kh   = (tid & 1) << 4;    // 0|16
  const float* xg = x + (size_t)(m0 + row2) * D_MODEL + kh;
  const u16t*  wg = W1b + (size_t)(tid >> 2) * D_MODEL + ((tid & 3) << 3);

  v8f acc[2][2];
#pragma unroll
  for (int i = 0; i < 2; ++i)
#pragma unroll
    for (int j = 0; j < 2; ++j) acc[i][j] = vzero8();

  vf4 xr[4];
  vu4 wr;
  {
    const vf4* xp = (const vf4*)xg;
#pragma unroll
    for (int i = 0; i < 4; ++i) xr[i] = xp[i];
    wr = *(const vu4*)wg;
  }
  // prologue: fill buffer 0
  {
    u16t* As = smem;
    u16t* Ws = smem + 4096;
    vu4 p0 = {packbf2(xr[0].x, xr[0].y), packbf2(xr[0].z, xr[0].w),
              packbf2(xr[1].x, xr[1].y), packbf2(xr[1].z, xr[1].w)};
    vu4 p1 = {packbf2(xr[2].x, xr[2].y), packbf2(xr[2].z, xr[2].w),
              packbf2(xr[3].x, xr[3].y), packbf2(xr[3].z, xr[3].w)};
    vu4* ad = (vu4*)(As + row2 * 32 + kh);
    ad[0] = p0; ad[1] = p1;
    *(vu4*)(Ws + (tid >> 2) * 32 + ((tid & 3) << 3)) = wr;
  }
  __syncthreads();

  for (int kt = 0; kt < D_MODEL / 32; ++kt) {
    const int  cur  = kt & 1;
    const bool more = (kt + 1) < (D_MODEL / 32);
    if (more) {  // issue next-tile global loads; they fly under the WMMAs
      xg += 32; wg += 32;
      const vf4* xp = (const vf4*)xg;
#pragma unroll
      for (int i = 0; i < 4; ++i) xr[i] = xp[i];
      wr = *(const vu4*)wg;
      __builtin_prefetch(xg + 32, 0, 3);
    }
    {
      const u16t* As = smem + cur * 6144;
      const u16t* Ws = As + 4096;
      v16bf af0 = load_frag(As, wm * 32 + (lane & 15),      32, lane, 0);
      v16bf af1 = load_frag(As, wm * 32 + 16 + (lane & 15), 32, lane, 0);
#pragma unroll
      for (int ni = 0; ni < 2; ++ni) {
        v16bf bf = load_frag(Ws, wn * 32 + ni * 16 + (lane & 15), 32, lane, 0);
        acc[0][ni] = wmma_bf16(af0, bf, acc[0][ni]);
        acc[1][ni] = wmma_bf16(af1, bf, acc[1][ni]);
      }
    }
    if (more) {  // store next tile into the other buffer
      u16t* As = smem + (cur ^ 1) * 6144;
      u16t* Ws = As + 4096;
      vu4 p0 = {packbf2(xr[0].x, xr[0].y), packbf2(xr[0].z, xr[0].w),
                packbf2(xr[1].x, xr[1].y), packbf2(xr[1].z, xr[1].w)};
      vu4 p1 = {packbf2(xr[2].x, xr[2].y), packbf2(xr[2].z, xr[2].w),
                packbf2(xr[3].x, xr[3].y), packbf2(xr[3].z, xr[3].w)};
      vu4* ad = (vu4*)(As + row2 * 32 + kh);
      ad[0] = p0; ad[1] = p1;
      *(vu4*)(Ws + (tid >> 2) * 32 + ((tid & 3) << 3)) = wr;
    }
    __syncthreads();
  }

  // epilogue: + b1, exact gelu, -> bf16
  const int ln = lane & 15;
  const int lh = ((lane >> 4) & 1) << 3;
#pragma unroll
  for (int mi = 0; mi < 2; ++mi)
#pragma unroll
    for (int ni = 0; ni < 2; ++ni) {
      int ng  = wn * 32 + ni * 16 + ln;
      float bv = b1[ng];
      int mb  = m0 + wm * 32 + mi * 16 + lh;
#pragma unroll
      for (int i = 0; i < 8; ++i) {
        float v = acc[mi][ni][i] + bv;
        float g = 0.5f * v * (1.0f + erff(v * 0.70710678118654752f));
        hout[(size_t)(mb + i) * DT_RANK + ng] = f2bf(g);
      }
    }
}

// ---------------------------------------------------------------------------
// Kernel 4: out = (x @ S) * (h @ W2^T + b2) + Dp
// tile 128x128, 256 threads = 8 waves (4x2), wave = 32x64 (2x4 WMMA subtiles)
// ping-pong LDS (2 x (As 8KB + Bs 8KB) = 32KB), one barrier per k-step;
// epilogue reuses the full 32KB for the fused K=64 dt GEMM.
// ---------------------------------------------------------------------------
__global__ __launch_bounds__(256) void SSM_main_gemm(
    const float* __restrict__ x,     // [8192][2048] fp32
    const u16t*  __restrict__ St,    // [2048][2048] bf16, St[n][k]
    const u16t*  __restrict__ hbf,   // [8192][64]   bf16
    const u16t*  __restrict__ W2b,   // [2048][64]   bf16, W2[n][r]
    const float* __restrict__ b2,    // [2048]
    const float* __restrict__ Dp,    // [2048]
    float*       __restrict__ out)   // [8192][2048]
{
  __shared__ __align__(16) u16t smem[16384];  // 32 KB total

  const int tid  = threadIdx.x;
  const int lane = tid & 31;
  const int wm   = (tid >> 5) >> 1;  // 0..3
  const int wn   = (tid >> 5) & 1;   // 0..1
  const int m0   = blockIdx.y * 128;
  const int n0   = blockIdx.x * 128;

  const int row2 = tid >> 1;        // 0..127
  const int kh   = (tid & 1) << 4;  // 0|16

  const float* xg = x  + (size_t)(m0 + row2) * D_MODEL + kh;
  const u16t*  sg = St + (size_t)(n0 + row2) * D_MODEL + kh;

  v8f acc[2][4];
#pragma unroll
  for (int i = 0; i < 2; ++i)
#pragma unroll
    for (int j = 0; j < 4; ++j) acc[i][j] = vzero8();

  vf4 xr[4];
  vu4 sr[2];
  {
    const vf4* xp = (const vf4*)xg;
#pragma unroll
    for (int i = 0; i < 4; ++i) xr[i] = xp[i];
    const vu4* sp = (const vu4*)sg;
    sr[0] = sp[0]; sr[1] = sp[1];
  }
  // prologue: fill buffer 0  (buf b: As = smem + b*8192, Bs = As + 4096)
  {
    u16t* As = smem;
    u16t* Bs = smem + 4096;
    vu4 p0 = {packbf2(xr[0].x, xr[0].y), packbf2(xr[0].z, xr[0].w),
              packbf2(xr[1].x, xr[1].y), packbf2(xr[1].z, xr[1].w)};
    vu4 p1 = {packbf2(xr[2].x, xr[2].y), packbf2(xr[2].z, xr[2].w),
              packbf2(xr[3].x, xr[3].y), packbf2(xr[3].z, xr[3].w)};
    vu4* ad = (vu4*)(As + row2 * 32 + kh);
    ad[0] = p0; ad[1] = p1;
    vu4* bd = (vu4*)(Bs + row2 * 32 + kh);
    bd[0] = sr[0]; bd[1] = sr[1];
  }
  __syncthreads();

  for (int kt = 0; kt < D_MODEL / 32; ++kt) {
    const int  cur  = kt & 1;
    const bool more = (kt + 1) < (D_MODEL / 32);
    if (more) {  // issue next-tile global loads; they fly under the WMMAs
      xg += 32; sg += 32;
      const vf4* xp = (const vf4*)xg;
#pragma unroll
      for (int i = 0; i < 4; ++i) xr[i] = xp[i];
      const vu4* sp = (const vu4*)sg;
      sr[0] = sp[0]; sr[1] = sp[1];
      __builtin_prefetch(xg + 32, 0, 3);
      __builtin_prefetch(sg + 32, 0, 3);
    }
    {
      const u16t* As = smem + cur * 8192;
      const u16t* Bs = As + 4096;
      v16bf af0 = load_frag(As, wm * 32 + (lane & 15),      32, lane, 0);
      v16bf af1 = load_frag(As, wm * 32 + 16 + (lane & 15), 32, lane, 0);
#pragma unroll
      for (int ni = 0; ni < 4; ++ni) {
        v16bf bf = load_frag(Bs, wn * 64 + ni * 16 + (lane & 15), 32, lane, 0);
        acc[0][ni] = wmma_bf16(af0, bf, acc[0][ni]);
        acc[1][ni] = wmma_bf16(af1, bf, acc[1][ni]);
      }
    }
    if (more) {  // store next tile into the other buffer
      u16t* As = smem + (cur ^ 1) * 8192;
      u16t* Bs = As + 4096;
      vu4 p0 = {packbf2(xr[0].x, xr[0].y), packbf2(xr[0].z, xr[0].w),
                packbf2(xr[1].x, xr[1].y), packbf2(xr[1].z, xr[1].w)};
      vu4 p1 = {packbf2(xr[2].x, xr[2].y), packbf2(xr[2].z, xr[2].w),
                packbf2(xr[3].x, xr[3].y), packbf2(xr[3].z, xr[3].w)};
      vu4* ad = (vu4*)(As + row2 * 32 + kh);
      ad[0] = p0; ad[1] = p1;
      vu4* bd = (vu4*)(Bs + row2 * 32 + kh);
      bd[0] = sr[0]; bd[1] = sr[1];
    }
    __syncthreads();
  }

  // ---- fused dt epilogue: dt = h_tile @ W2_tile^T + b2 (K = 64, 2 WMMA steps)
  {
    // Hs = smem[0..8192): [128][64] bf16 ; W2s = smem[8192..16384): [128][64]
    const int r2  = tid >> 1;
    const int c32 = (tid & 1) << 5;  // 0|32 halves
    const vu4* hp = (const vu4*)(hbf + (size_t)(m0 + r2) * DT_RANK + c32);
    vu4* hd = (vu4*)(smem + r2 * 64 + c32);
#pragma unroll
    for (int i = 0; i < 4; ++i) hd[i] = hp[i];
    const vu4* wp = (const vu4*)(W2b + (size_t)(n0 + r2) * DT_RANK + c32);
    vu4* wd = (vu4*)(smem + 8192 + r2 * 64 + c32);
#pragma unroll
    for (int i = 0; i < 4; ++i) wd[i] = wp[i];
  }
  __syncthreads();

  v8f dacc[2][4];
#pragma unroll
  for (int i = 0; i < 2; ++i)
#pragma unroll
    for (int j = 0; j < 4; ++j) dacc[i][j] = vzero8();

#pragma unroll
  for (int ks = 0; ks < 2; ++ks) {
    const int ko = ks * 32;
    v16bf af0 = load_frag(smem, wm * 32 + (lane & 15),      64, lane, ko);
    v16bf af1 = load_frag(smem, wm * 32 + 16 + (lane & 15), 64, lane, ko);
#pragma unroll
    for (int ni = 0; ni < 4; ++ni) {
      v16bf bf = load_frag(smem + 8192, wn * 64 + ni * 16 + (lane & 15), 64, lane, ko);
      dacc[0][ni] = wmma_bf16(af0, bf, dacc[0][ni]);
      dacc[1][ni] = wmma_bf16(af1, bf, dacc[1][ni]);
    }
  }

  // ---- final: out = acc * (dacc + b2) + Dp
  const int ln = lane & 15;
  const int lh = ((lane >> 4) & 1) << 3;
#pragma unroll
  for (int mi = 0; mi < 2; ++mi)
#pragma unroll
    for (int ni = 0; ni < 4; ++ni) {
      int   ng  = n0 + wn * 64 + ni * 16 + ln;
      float b2v = b2[ng];
      float dpv = Dp[ng];
      int   mb  = m0 + wm * 32 + mi * 16 + lh;
#pragma unroll
      for (int i = 0; i < 8; ++i) {
        float dt = dacc[mi][ni][i] + b2v;
        out[(size_t)(mb + i) * D_MODEL + ng] = acc[mi][ni][i] * dt + dpv;
      }
    }
}

// ---------------------------------------------------------------------------
// Launch
// ---------------------------------------------------------------------------
extern "C" void kernel_launch(void* const* d_in, const int* in_sizes, int n_in,
                              void* d_out, int out_size, void* d_ws, size_t ws_size,
                              hipStream_t stream) {
  (void)in_sizes; (void)n_in; (void)out_size; (void)ws_size;
  const float* x  = (const float*)d_in[0];
  const float* A  = (const float*)d_in[1];
  const float* B  = (const float*)d_in[2];
  const float* C  = (const float*)d_in[3];
  const float* Dp = (const float*)d_in[4];
  const float* W1 = (const float*)d_in[5];
  const float* b1 = (const float*)d_in[6];
  const float* W2 = (const float*)d_in[7];
  const float* b2 = (const float*)d_in[8];
  float* out = (float*)d_out;

  // workspace layout (bytes): St 8MB | W1b 256KB | W2b 256KB | hbf 1MB
  char* ws   = (char*)d_ws;
  u16t* St   = (u16t*)ws;
  u16t* W1b  = (u16t*)(ws + (size_t)D_MODEL * D_MODEL * 2);
  u16t* W2b  = W1b + (size_t)DT_RANK * D_MODEL;
  u16t* hbf  = W2b + (size_t)D_MODEL * DT_RANK;

  const int nw = DT_RANK * D_MODEL;  // 131072
  SSM_conv_bf16<<<(nw + 255) / 256, 256, 0, stream>>>(W1, W1b, nw);
  SSM_conv_bf16<<<(nw + 255) / 256, 256, 0, stream>>>(W2, W2b, nw);
  SSM_sum_transpose<<<dim3(64, 64), dim3(32, 8), 0, stream>>>(A, B, C, St);
  SSM_h_gemm<<<M_ROWS / 128, 256, 0, stream>>>(x, W1b, b1, hbf);
  SSM_main_gemm<<<dim3(D_MODEL / 128, M_ROWS / 128), 256, 0, stream>>>(
      x, St, hbf, W2b, b2, Dp, out);
}
